// WithGNN_encoder_90426241450347
// MI455X (gfx1250) — matrix operand order repaired and checked by reading, not compiled, not run on hardware
//
#include <hip/hip_runtime.h>
#include <hip/hip_bf16.h>

// ---------------------------------------------------------------------------
// CDNA5 (gfx1250) GNN encoder.
//  - All dense node GEMMs via v_wmma_f32_16x16x32_bf16, register-blocked NT
//    output tiles per wave, LDS-staged fragment-ready bf16 weights.
//  - Fused edge message kernel with analytic bond-BN (16x16 moment matrix),
//    f32 global atomics for scatter-add, async-to-LDS weight staging via
//    global_load_async_to_lds_b128 (signature: v4i AS1*, v4i AS3*, imm, imm).
// ---------------------------------------------------------------------------

typedef __bf16 v16bf __attribute__((ext_vector_type(16)));
typedef float  v8f   __attribute__((ext_vector_type(8)));
typedef int    v4i_  __attribute__((ext_vector_type(4)));

typedef __attribute__((address_space(1))) v4i_ as1_v4i;  // global int4
typedef __attribute__((address_space(3))) v4i_ as3_v4i;  // LDS int4

#if defined(__has_builtin)
# if __has_builtin(__builtin_amdgcn_global_load_async_to_lds_b128)
#  define HAVE_ASYNC_LDS 1
# endif
#endif
#ifndef HAVE_ASYNC_LDS
# define HAVE_ASYNC_LDS 0
#endif

#define NF   9
#define CARD 16
#define D    64
#define BF_  16

// ------------------------------- utility -----------------------------------

__global__ void fill_f32(float* __restrict__ p, long n, float v) {
  long i = (long)blockIdx.x * blockDim.x + threadIdx.x;
  if (i < n) p[i] = v;
}

__global__ void count_deg(const int* __restrict__ row, float* __restrict__ deg, int E) {
  int e = blockIdx.x * blockDim.x + threadIdx.x;
  if (e < E) atomicAdd(&deg[row[e]], 1.0f);
}

__global__ void dinv_kernel(const float* __restrict__ deg, float* __restrict__ dinv, int N) {
  int i = blockIdx.x * blockDim.x + threadIdx.x;
  if (i < N) dinv[i] = rsqrtf(deg[i]);
}

// --------------------- atom encoder: concat buffer --------------------------
__global__ void atom_concat(const float* __restrict__ atom_x,
                            const int* __restrict__ feat,
                            const float* __restrict__ emb,
                            float* __restrict__ Ac, int N) {
  long i = (long)blockIdx.x * blockDim.x + threadIdx.x;
  if (i >= (long)N * 128) return;
  int n = (int)(i >> 7), d = (int)(i & 127);
  if (d < D) {
    Ac[i] = atom_x[(size_t)n * D + d];
  } else {
    int dd = d - D;
    float s = 0.f;
#pragma unroll
    for (int f = 0; f < NF; ++f) {
      int c = feat[n * NF + f];
      s += emb[((size_t)(f * CARD + c)) * D + dd];
    }
    Ac[i] = s;
  }
}

// ----------------------- WMMA bf16 GEMM -------------------------------------
// C[M x (NT*16)] = A[M x K] @ B + bias, B(k,n) = W[k*sk + n*sn].
// Block = 4 waves; each wave owns a 16-row stripe and all NT column tiles.
// W is staged once per block into LDS in fragment-ready bf16 layout:
//   frag(kt, nt) = 32 lanes x 16 bf16 (32B per lane) -> one wide ds_load.

template <int NT>
__global__ __launch_bounds__(128) void wmma_gemm_bias(
    const float* __restrict__ A, const float* __restrict__ W,
    const float* __restrict__ bias, float* __restrict__ C,
    int M, int K, int sk, int sn) {
  constexpr int Nout = NT * 16;
  extern __shared__ char smem[];
  v16bf* BW = (v16bf*)smem;  // (K/32)*NT*32 fragments of 32B

  const int tid  = threadIdx.x;
  const int lane = tid & 31;
  const int wave = tid >> 5;
  const int KT   = K >> 5;

  // ---- stage W -> LDS (each W element read exactly once per block) ----
  for (int slot = tid; slot < KT * NT * 32; slot += 128) {
    int l  = slot & 31;
    int nt = (slot >> 5) % NT;
    int kt = (slot >> 5) / NT;
    int n  = nt * 16 + (l & 15);
    int kb = kt * 32 + ((l & 16) ? 16 : 0);
    v16bf b;
#pragma unroll
    for (int j = 0; j < 16; ++j)
      b[j] = (__bf16)W[(size_t)(kb + j) * sk + (size_t)n * sn];
    BW[slot] = b;
  }
  __syncthreads();

  const int m0 = (blockIdx.x * 4 + wave) * 16;
  if (m0 >= M) return;  // whole-wave exit: EXEC stays all-ones for WMMA

  v8f acc[NT];
#pragma unroll
  for (int t = 0; t < NT; ++t) acc[t] = (v8f){};

  const int mrow = m0 + (lane & 15);
  const int koff = (lane & 16) ? 16 : 0;  // contiguous k: upper half-wave at K+16

  for (int kt = 0; kt < KT; ++kt) {
    const float* ap = A + (size_t)mrow * K + kt * 32 + koff;
    float4 a0 = *(const float4*)(ap);
    float4 a1 = *(const float4*)(ap + 4);
    float4 a2 = *(const float4*)(ap + 8);
    float4 a3 = *(const float4*)(ap + 12);
    v16bf a;
    a[0]  = (__bf16)a0.x; a[1]  = (__bf16)a0.y; a[2]  = (__bf16)a0.z; a[3]  = (__bf16)a0.w;
    a[4]  = (__bf16)a1.x; a[5]  = (__bf16)a1.y; a[6]  = (__bf16)a1.z; a[7]  = (__bf16)a1.w;
    a[8]  = (__bf16)a2.x; a[9]  = (__bf16)a2.y; a[10] = (__bf16)a2.z; a[11] = (__bf16)a2.w;
    a[12] = (__bf16)a3.x; a[13] = (__bf16)a3.y; a[14] = (__bf16)a3.z; a[15] = (__bf16)a3.w;
#pragma unroll
    for (int nt = 0; nt < NT; ++nt) {
      v16bf b = BW[(kt * NT + nt) * 32 + lane];
      acc[nt] = __builtin_amdgcn_wmma_f32_16x16x32_bf16(false, a, false, b,
                                                        (short)0, acc[nt], false, false);
    }
  }

  const int cn_l  = lane & 15;
  const int rbase = (lane & 16) ? 8 : 0;
#pragma unroll
  for (int nt = 0; nt < NT; ++nt) {
    int cn = nt * 16 + cn_l;
    float bv = bias[cn];
#pragma unroll
    for (int r = 0; r < 8; ++r)
      C[(size_t)(m0 + rbase + r) * Nout + cn] = acc[nt][r] + bv;
  }
}

// ------------------- edge_attr moments: mu (16), S (16x16) ------------------
__global__ void edge_stats(const float* __restrict__ ea,
                           float* __restrict__ mu_sum,
                           float* __restrict__ S_sum, int E) {
  __shared__ float tile[256 * BF_];
  int i = threadIdx.x >> 4;
  int j = threadIdx.x & 15;
  float sacc = 0.f, macc = 0.f;
  for (int base = blockIdx.x * 256; base < E; base += gridDim.x * 256) {
    int cnt = min(256, E - base);
    __syncthreads();
    for (int t = threadIdx.x; t < cnt * BF_; t += 256)
      tile[t] = ea[(size_t)base * BF_ + t];
    __syncthreads();
    for (int e = 0; e < cnt; ++e) {
      float ai = tile[e * BF_ + i];
      float aj = tile[e * BF_ + j];
      sacc += ai * aj;
      if (i == 0) macc += aj;
    }
  }
  atomicAdd(&S_sum[i * BF_ + j], sacc);
  if (i == 0) atomicAdd(&mu_sum[j], macc);
}

// ---- bond-BN folded into per-column affine: e_bn = (a.w_d)*scale + shift ---
__global__ void bond_finalize(const float* __restrict__ mu_sum,
                              const float* __restrict__ S_sum,
                              const float* __restrict__ bondW,
                              const float* __restrict__ bondb,
                              const float* __restrict__ g,
                              const float* __restrict__ beta,
                              float* __restrict__ escale,
                              float* __restrict__ eshift, float invE) {
  int d = threadIdx.x;  // 64 threads
  float w[BF_], mu[BF_];
#pragma unroll
  for (int k = 0; k < BF_; ++k) {
    w[k]  = bondW[k * D + d];
    mu[k] = mu_sum[k] * invE;
  }
  float bd = bondb[d];
  float wmu = 0.f;
#pragma unroll
  for (int k = 0; k < BF_; ++k) wmu += w[k] * mu[k];
  float mean = wmu + bd;
  float ex2 = 0.f;
#pragma unroll
  for (int k = 0; k < BF_; ++k) {
    float t = 0.f;
#pragma unroll
    for (int k2 = 0; k2 < BF_; ++k2) t += (S_sum[k * BF_ + k2] * invE) * w[k2];
    ex2 += w[k] * t;
  }
  ex2 += 2.f * bd * wmu + bd * bd;
  float var = ex2 - mean * mean;
  float s = g[d] * rsqrtf(var + 1e-6f);
  escale[d] = s;
  eshift[d] = beta[d] - mean * s;
}

// ----------------- fused message + scatter-add (64 thr/edge) ----------------
__global__ __launch_bounds__(256) void message_kernel(
    const int* __restrict__ row, const int* __restrict__ col,
    const float* __restrict__ edge_attr, const float* __restrict__ x,
    const float* __restrict__ dinv, const float* __restrict__ bondW,
    const float* __restrict__ escale, const float* __restrict__ eshift,
    float* __restrict__ aggr, int E) {
  __shared__ __align__(16) float Ws[BF_ * D];   // 4KB
  __shared__ float sc[D], sh[D];

  // bulk 4KB copy: gfx1250 async-to-LDS when available (one b128 per thread)
#if HAVE_ASYNC_LDS
  __builtin_amdgcn_global_load_async_to_lds_b128(
      (as1_v4i*)(bondW + threadIdx.x * 4),
      (as3_v4i*)(&Ws[threadIdx.x * 4]), 0, 0);
#else
  *(float4*)&Ws[threadIdx.x * 4] = *(const float4*)(bondW + threadIdx.x * 4);
#endif
  if (threadIdx.x < D) {
    sc[threadIdx.x] = escale[threadIdx.x];
    sh[threadIdx.x] = eshift[threadIdx.x];
  }
#if HAVE_ASYNC_LDS
# if __has_builtin(__builtin_amdgcn_s_wait_asynccnt)
  __builtin_amdgcn_s_wait_asynccnt(0);
# else
  asm volatile("s_wait_asynccnt 0x0" ::: "memory");
# endif
#endif
  __syncthreads();

  int d    = threadIdx.x & (D - 1);
  int eIdx = blockIdx.x * (256 / D) + (threadIdx.x / D);
  if (eIdx >= E) return;

  if (d == 0 && eIdx + 256 < E)
    __builtin_prefetch(&edge_attr[(size_t)(eIdx + 256) * BF_], 0, 0);

  int r = row[eIdx], c = col[eIdx];
  float norm = dinv[r] * dinv[c];
  const float* a = edge_attr + (size_t)eIdx * BF_;
  float acc = 0.f;
#pragma unroll
  for (int k = 0; k < BF_; ++k) acc = fmaf(a[k], Ws[k * D + d], acc);
  float e  = acc * sc[d] + sh[d];
  float xv = x[(size_t)r * D + d];
  float m  = fmaxf(xv + e, 0.f) * norm;
  atomicAdd(&aggr[(size_t)c * D + d], m);
}

// --------------------------- GRU + root/self term ---------------------------
__global__ void gru_update(const float* __restrict__ gi, const float* __restrict__ gh,
                           const float* __restrict__ x, const float* __restrict__ root,
                           const float* __restrict__ deg, float* __restrict__ hl, int N) {
  long i = (long)blockIdx.x * blockDim.x + threadIdx.x;
  if (i >= (long)N * D) return;
  int n = (int)(i >> 6), d = (int)(i & 63);
  const float* gin = gi + (size_t)n * 192;
  const float* ghn = gh + (size_t)n * 192;
  float r  = 1.f / (1.f + expf(-(gin[d]      + ghn[d])));
  float z  = 1.f / (1.f + expf(-(gin[64 + d] + ghn[64 + d])));
  float nn = tanhf(gin[128 + d] + r * ghn[128 + d]);
  float xv = x[i];
  float upd = (1.f - z) * nn + z * xv;
  hl[i] = upd + fmaxf(xv + root[d], 0.f) / deg[n];
}

// ------------------------- output BN (reduce/finalize/apply) ----------------
__global__ void bn_reduce(const float* __restrict__ v, float* __restrict__ sums,
                          float* __restrict__ sumsq, int N) {
  int d = threadIdx.x & (D - 1);
  int sub = threadIdx.x >> 6;
  float s = 0.f, s2 = 0.f;
  for (int n = blockIdx.x * 4 + sub; n < N; n += gridDim.x * 4) {
    float val = v[(size_t)n * D + d];
    s += val; s2 += val * val;
  }
  __shared__ float ls[256], ls2[256];
  ls[threadIdx.x] = s; ls2[threadIdx.x] = s2;
  __syncthreads();
  if (sub == 0) {
    s  = ls[d]  + ls[64 + d]  + ls[128 + d]  + ls[192 + d];
    s2 = ls2[d] + ls2[64 + d] + ls2[128 + d] + ls2[192 + d];
    atomicAdd(&sums[d], s);
    atomicAdd(&sumsq[d], s2);
  }
}

__global__ void bn_finalize(const float* __restrict__ sums, const float* __restrict__ sumsq,
                            const float* __restrict__ g, const float* __restrict__ b,
                            float* __restrict__ sc, float* __restrict__ sh,
                            float invN, float eps) {
  int d = threadIdx.x;  // 64
  float mean = sums[d] * invN;
  float var  = sumsq[d] * invN - mean * mean;
  float s = g[d] * rsqrtf(var + eps);
  sc[d] = s;
  sh[d] = b[d] - mean * s;
}

__global__ void bn_apply(const float* __restrict__ v, const float* __restrict__ sc,
                         const float* __restrict__ sh, float* __restrict__ out,
                         int N, int relu) {
  long i = (long)blockIdx.x * blockDim.x + threadIdx.x;
  if (i >= (long)N * D) return;
  int d = (int)(i & 63);
  float o = v[i] * sc[d] + sh[d];
  out[i] = relu ? fmaxf(o, 0.f) : o;
}

// ------------------------------- host driver --------------------------------

extern "C" void kernel_launch(void* const* d_in, const int* in_sizes, int n_in,
                              void* d_out, int out_size, void* d_ws, size_t ws_size,
                              hipStream_t stream) {
  const float* atom_x   = (const float*)d_in[0];
  const int*   atom_ft  = (const int*)  d_in[1];
  const int*   ei       = (const int*)  d_in[2];
  const float* edge_at  = (const float*)d_in[3];
  const float* atom_emb = (const float*)d_in[4];
  const float* proj_W   = (const float*)d_in[5];
  const float* proj_b   = (const float*)d_in[6];
  const float* lin_W    = (const float*)d_in[7];
  const float* lin_b    = (const float*)d_in[8];
  const float* root_emb = (const float*)d_in[9];
  const float* bond_W   = (const float*)d_in[10];
  const float* bond_b   = (const float*)d_in[11];
  const float* bond_g   = (const float*)d_in[12];
  const float* bond_bt  = (const float*)d_in[13];
  const float* gru_Wih  = (const float*)d_in[14];
  const float* gru_bih  = (const float*)d_in[15];
  const float* gru_Whh  = (const float*)d_in[16];
  const float* gru_bhh  = (const float*)d_in[17];
  const float* bn_g     = (const float*)d_in[18];
  const float* bn_b     = (const float*)d_in[19];

  const int N = in_sizes[0] / D;
  const int E = in_sizes[2] / 2;
  const int* row = ei;
  const int* col = ei + E;

  // workspace carve-out
  float* ws = (float*)d_ws;
  size_t off = 0;
  float* deg   = ws + off; off += N;
  float* dinv  = ws + off; off += N;
  float* Ac    = ws + off; off += (size_t)N * 128;  // concat; aliased as hl later
  float* h     = ws + off; off += (size_t)N * D;
  float* x     = ws + off; off += (size_t)N * D;
  float* aggr  = ws + off; off += (size_t)N * D;
  float* gi    = ws + off; off += (size_t)N * 192;
  float* gh    = ws + off; off += (size_t)N * 192;
  float* musum = ws + off; off += 16;
  float* Ssum  = ws + off; off += 256;
  float* escl  = ws + off; off += 64;
  float* eshf  = ws + off; off += 64;
  float* bnsum = ws + off; off += 64;
  float* bnsq  = ws + off; off += 64;
  float* bnsc  = ws + off; off += 64;
  float* bnsh  = ws + off; off += 64;
  float* hl    = Ac;  // Ac is dead after the proj GEMM

  const int T = 256;
  auto blocks = [](long n, int t) { return (unsigned)((n + t - 1) / t); };
  const unsigned mblk = (unsigned)((N / 16 + 3) / 4);  // 4 waves (64 rows) per block

  // degree (deg = segment_sum(ones,row) + 1), dinv
  fill_f32<<<dim3(blocks(N, T)), dim3(T), 0, stream>>>(deg, (long)N, 1.0f);
  count_deg<<<dim3(blocks(E, T)), dim3(T), 0, stream>>>(row, deg, E);
  dinv_kernel<<<dim3(blocks(N, T)), dim3(T), 0, stream>>>(deg, dinv, N);

  // edge_attr moments (layer-independent)
  fill_f32<<<dim3(2), dim3(T), 0, stream>>>(musum, (long)(16 + 256), 0.0f);
  edge_stats<<<dim3(512), dim3(256), 0, stream>>>(edge_at, musum, Ssum, E);

  // atom fusion encoder: concat + proj GEMM (WMMA, K=128, NT=4 -> 16KB LDS)
  atom_concat<<<dim3(blocks((long)N * 128, T)), dim3(T), 0, stream>>>(
      atom_x, atom_ft, atom_emb, Ac, N);
  wmma_gemm_bias<4><<<dim3(mblk), dim3(128), 4 * 4 * 1024, stream>>>(
      Ac, proj_W, proj_b, h, N, 2 * D, D, 1);

  const float invE = 1.0f / (float)E;
  const float invN = 1.0f / (float)N;

  for (int l = 0; l < 3; ++l) {
    const float* lW  = lin_W   + (size_t)l * D * D;
    const float* lb  = lin_b   + (size_t)l * D;
    const float* rt  = root_emb+ (size_t)l * D;
    const float* bW  = bond_W  + (size_t)l * BF_ * D;
    const float* bb  = bond_b  + (size_t)l * D;
    const float* bg  = bond_g  + (size_t)l * D;
    const float* bbt = bond_bt + (size_t)l * D;
    const float* Wih = gru_Wih + (size_t)l * 192 * D;
    const float* bih = gru_bih + (size_t)l * 192;
    const float* Whh = gru_Whh + (size_t)l * 192 * D;
    const float* bhh = gru_bhh + (size_t)l * 192;
    const float* og  = bn_g    + (size_t)l * D;
    const float* ob  = bn_b    + (size_t)l * D;

    // x = h @ lin_W + lin_b   (K=64, NT=4 -> 8KB LDS)
    wmma_gemm_bias<4><<<dim3(mblk), dim3(128), 2 * 4 * 1024, stream>>>(
        h, lW, lb, x, N, D, D, 1);

    // bond BN -> per-column affine
    bond_finalize<<<dim3(1), dim3(64), 0, stream>>>(
        musum, Ssum, bW, bb, bg, bbt, escl, eshf, invE);

    // scatter-add messages
    fill_f32<<<dim3(blocks((long)N * D, T)), dim3(T), 0, stream>>>(
        aggr, (long)N * D, 0.0f);
    message_kernel<<<dim3(blocks(E, T / D)), dim3(T), 0, stream>>>(
        row, col, edge_at, x, dinv, bW, escl, eshf, aggr, E);

    // GRU gate GEMMs: gi = aggr @ Wih^T ; gh = x @ Whh^T  (K=64, NT=12 -> 24KB LDS)
    wmma_gemm_bias<12><<<dim3(mblk), dim3(128), 2 * 12 * 1024, stream>>>(
        aggr, Wih, bih, gi, N, D, 1, D);
    wmma_gemm_bias<12><<<dim3(mblk), dim3(128), 2 * 12 * 1024, stream>>>(
        x, Whh, bhh, gh, N, D, 1, D);

    // gates + root/self term
    gru_update<<<dim3(blocks((long)N * D, T)), dim3(T), 0, stream>>>(
        gi, gh, x, rt, deg, hl, N);

    // output BN (+ReLU except last layer)
    fill_f32<<<dim3(1), dim3(T), 0, stream>>>(bnsum, (long)128, 0.0f);
    bn_reduce<<<dim3(256), dim3(256), 0, stream>>>(hl, bnsum, bnsq, N);
    bn_finalize<<<dim3(1), dim3(64), 0, stream>>>(
        bnsum, bnsq, og, ob, bnsc, bnsh, invN, 1e-5f);
    float* dst = (l == 2) ? (float*)d_out : h;
    bn_apply<<<dim3(blocks((long)N * D, T)), dim3(T), 0, stream>>>(
        hl, bnsc, bnsh, dst, N, (l < 2) ? 1 : 0);
  }
}
